// GATConv_77541339562447
// MI455X (gfx1250) — matrix operand order repaired and checked by reading, not compiled, not run on hardware
//
#include <hip/hip_runtime.h>

// ---------------- problem constants (from reference) ----------------
#define N_NODES 100000
#define N_EDGES 1600000
#define IN_CH   128
#define HEADS   4
#define CPH     16        // channels per head
#define HC      64        // HEADS*CPH
#define SLOPE   0.2f

typedef __attribute__((ext_vector_type(16))) _Float16 v16h;
typedef __attribute__((ext_vector_type(8)))  _Float16 v8h;
typedef __attribute__((ext_vector_type(8)))  float    v8f;

// ---- ordered-int encoding of float: monotonic, self-inverse ----
__device__ __forceinline__ int f2ord(float f) {
    int i = __float_as_int(f);
    return (i < 0) ? (i ^ 0x7fffffff) : i;
}
__device__ __forceinline__ float ord2f(int i) {
    return __int_as_float((i < 0) ? (i ^ 0x7fffffff) : i);
}

// =====================================================================
// K1: convert x (f32 -> f16, row major [N,128]) and build transposed
//     f16 weight wT[64][128] so WMMA B fragments load contiguous K.
// =====================================================================
__global__ void k_convert(const float* __restrict__ x,
                          const float* __restrict__ w,
                          _Float16* __restrict__ xh,
                          _Float16* __restrict__ wT) {
    int t = blockIdx.x * blockDim.x + threadIdx.x;
    if (t < N_NODES * IN_CH) xh[t] = (_Float16)x[t];
    if (t < IN_CH * HC) {             // w row-major [128][64]
        int k = t / HC, c = t % HC;
        wT[c * IN_CH + k] = (_Float16)w[t];
    }
}

// =====================================================================
// K2: ft[N,64] = xh[N,128] * w[128,64] via v_wmma_f32_16x16x32_f16.
//     One wave -> 16 rows x 64 cols. 4 k-chunks x 4 col tiles = 16 WMMA.
//     Software-pipelined: A + 4 B fragments double-buffered so k+1 loads
//     overlap k's WMMAs (avoids full s_wait_loadcnt 0x0 drains).
// A 16x32 f16 as loaded here (A and B share this k order): lane<16: row=lane, K = k0..k0+15
//                     lane>=16: same row, K = k0+16..k0+31
// B 32x16 f16 as loaded here (same k order as A): lane&15 = column, lane<16: K=k0..k0+15,
//                     lane>=16: K=k0+16..k0+31 (contiguous in wT).
// C/D f32: VGPR i -> M = i + (lane>=16 ? 8 : 0), N = lane&15.
// =====================================================================
__global__ void k_gemm_wmma(const _Float16* __restrict__ xh,
                            const _Float16* __restrict__ wT,
                            float* __restrict__ ft) {
    const int lane = threadIdx.x & 31;
    const int wave = blockIdx.x * (blockDim.x >> 5) + (threadIdx.x >> 5);
    const int r0 = wave * 16;
    if (r0 >= N_NODES) return;

    const int mrow = lane & 15;
    const int sel  = lane >> 4;
    const _Float16* arow = xh + (size_t)(r0 + mrow) * IN_CH;

    auto loadA = [&](int k0) {
        v8h lo = *(const v8h*)(arow + k0 + sel * 16);
        v8h hi = *(const v8h*)(arow + k0 + sel * 16 + 8);
        v16h a;
#pragma unroll
        for (int i = 0; i < 8; ++i) { a[i] = lo[i]; a[i + 8] = hi[i]; }
        return a;
    };
    auto loadB = [&](int n, int k0) {
        const _Float16* bcol =
            wT + (size_t)(n * 16 + mrow) * IN_CH + k0 + sel * 16;
        v8h lo = *(const v8h*)(bcol);
        v8h hi = *(const v8h*)(bcol + 8);
        v16h b;
#pragma unroll
        for (int i = 0; i < 8; ++i) { b[i] = lo[i]; b[i + 8] = hi[i]; }
        return b;
    };

    v8f acc[4] = {v8f{}, v8f{}, v8f{}, v8f{}};

    // prologue: stage k-chunk 0
    v16h a_cur = loadA(0);
    v16h b_cur[4];
#pragma unroll
    for (int n = 0; n < 4; ++n) b_cur[n] = loadB(n, 0);

#pragma unroll
    for (int kc = 0; kc < 4; ++kc) {
        v16h a_nxt;
        v16h b_nxt[4];
        if (kc < 3) {                       // issue k+1 loads BEFORE k's WMMAs
            const int k1 = (kc + 1) * 32;
            a_nxt = loadA(k1);
#pragma unroll
            for (int n = 0; n < 4; ++n) b_nxt[n] = loadB(n, k1);
        }
#pragma unroll
        for (int n = 0; n < 4; ++n) {
            acc[n] = __builtin_amdgcn_wmma_f32_16x16x32_f16(
                false, a_cur, false, b_cur[n], (short)0, acc[n], false, false);
        }
        if (kc < 3) {
            a_cur = a_nxt;
#pragma unroll
            for (int n = 0; n < 4; ++n) b_cur[n] = b_nxt[n];
        }
    }

    const int mbase = sel * 8;
    const int ncol  = lane & 15;
#pragma unroll
    for (int n = 0; n < 4; ++n) {
        float* dstp = ft + (size_t)(r0 + mbase) * HC + n * 16 + ncol;
#pragma unroll
        for (int i = 0; i < 8; ++i) dstp[(size_t)i * HC] = acc[n][i];
    }
}

// =====================================================================
// K3: per-node attention logits a1[n,h], a2[n,h]
// =====================================================================
__global__ void k_attn(const float* __restrict__ ft,
                       const float* __restrict__ al,
                       const float* __restrict__ ar,
                       float* __restrict__ a1, float* __restrict__ a2) {
    int t = blockIdx.x * blockDim.x + threadIdx.x;
    if (t >= N_NODES * HEADS) return;
    int h = t & 3, n = t >> 2;
    const float* f = ft + (size_t)n * HC + h * CPH;
    float s1 = 0.f, s2 = 0.f;
#pragma unroll
    for (int c = 0; c < CPH; ++c) {
        float v = f[c];
        s1 += v * al[h * CPH + c];
        s2 += v * ar[h * CPH + c];
    }
    a1[t] = s1;
    a2[t] = s2;
}

// =====================================================================
// K4: init accumulators (d_out is the accumulation buffer)
// =====================================================================
__global__ void k_init(float* __restrict__ out, int* __restrict__ emax,
                       float* __restrict__ z) {
    int t = blockIdx.x * blockDim.x + threadIdx.x;
    if (t < N_NODES * HC) out[t] = 0.f;
    if (t < N_NODES * HEADS) { emax[t] = (int)0x80000000; z[t] = 0.f; }
}

// =====================================================================
// K5: segment max of leaky-relu edge logits via ordered-int atomicMax
// =====================================================================
__global__ void k_edge_max(const int* __restrict__ src,
                           const int* __restrict__ dst,
                           const float* __restrict__ a1,
                           const float* __restrict__ a2,
                           int* __restrict__ emax) {
    int t = blockIdx.x * blockDim.x + threadIdx.x;
    if (t >= N_EDGES * HEADS) return;
    int h = t & 3, e = t >> 2;
    int s = src[e], d = dst[e];
    float v = a1[s * 4 + h] + a2[d * 4 + h];
    v = (v >= 0.f) ? v : SLOPE * v;
    atomicMax(&emax[d * 4 + h], f2ord(v));
}

// =====================================================================
// K6: s = exp(e - max); z[dst,h] += s; out[dst,h,:] += s * ft[src,h,:]
// =====================================================================
__global__ void k_edge_acc(const int* __restrict__ src,
                           const int* __restrict__ dst,
                           const float* __restrict__ a1,
                           const float* __restrict__ a2,
                           const int* __restrict__ emax,
                           const float* __restrict__ ft,
                           float* __restrict__ z, float* __restrict__ out) {
    int t = blockIdx.x * blockDim.x + threadIdx.x;
    if (t >= N_EDGES * HEADS) return;
    int h = t & 3, e = t >> 2;
    int s = src[e], d = dst[e];
    float v = a1[s * 4 + h] + a2[d * 4 + h];
    v = (v >= 0.f) ? v : SLOPE * v;
    float m  = ord2f(emax[d * 4 + h]);
    float sv = __expf(v - m);
    atomicAdd(&z[d * 4 + h], sv);

    const float4* fsrc = (const float4*)(ft + (size_t)s * HC + h * CPH);
    float* o = out + (size_t)d * HC + h * CPH;
#pragma unroll
    for (int q = 0; q < 4; ++q) {
        float4 fv = fsrc[q];
        atomicAdd(o + q * 4 + 0, sv * fv.x);
        atomicAdd(o + q * 4 + 1, sv * fv.y);
        atomicAdd(o + q * 4 + 2, sv * fv.z);
        atomicAdd(o + q * 4 + 3, sv * fv.w);
    }
}

// =====================================================================
// K7: normalize by softmax partition z
// =====================================================================
__global__ void k_norm(float* __restrict__ out, const float* __restrict__ z) {
    int t = blockIdx.x * blockDim.x + threadIdx.x;
    if (t >= N_NODES * HC) return;
    int n = t / HC, h = (t % HC) / CPH;
    out[t] /= z[n * 4 + h];
}

// =====================================================================
extern "C" void kernel_launch(void* const* d_in, const int* in_sizes, int n_in,
                              void* d_out, int out_size, void* d_ws, size_t ws_size,
                              hipStream_t stream) {
    const float* x      = (const float*)d_in[0];
    const float* fc_w   = (const float*)d_in[1];
    const float* attn_l = (const float*)d_in[2];
    const float* attn_r = (const float*)d_in[3];
    const int*   src    = (const int*)d_in[4];
    const int*   dst    = (const int*)d_in[5];
    float* out = (float*)d_out;

    char* ws = (char*)d_ws;
    size_t off = 0;
    auto take = [&](size_t bytes) {
        void* p = ws + off;
        off = (off + bytes + 255) & ~(size_t)255;
        return p;
    };
    _Float16* xh = (_Float16*)take((size_t)N_NODES * IN_CH * sizeof(_Float16));
    _Float16* wT = (_Float16*)take((size_t)IN_CH * HC * sizeof(_Float16));
    float* ft    = (float*)take((size_t)N_NODES * HC * sizeof(float));
    float* a1    = (float*)take((size_t)N_NODES * HEADS * sizeof(float));
    float* a2    = (float*)take((size_t)N_NODES * HEADS * sizeof(float));
    int*   emax  = (int*)take((size_t)N_NODES * HEADS * sizeof(int));
    float* z     = (float*)take((size_t)N_NODES * HEADS * sizeof(float));
    (void)ws_size; (void)in_sizes; (void)n_in; (void)out_size;

    const int B = 256;
    int t;

    t = N_NODES * IN_CH;
    k_convert<<<(t + B - 1) / B, B, 0, stream>>>(x, fc_w, xh, wT);

    // one wave per 16-row tile; 8 waves per 256-thread block
    int waves  = N_NODES / 16;               // 6250 (N divisible by 16)
    int blocks = (waves + 7) / 8;
    k_gemm_wmma<<<blocks, B, 0, stream>>>(xh, wT, ft);

    t = N_NODES * HEADS;
    k_attn<<<(t + B - 1) / B, B, 0, stream>>>(ft, attn_l, attn_r, a1, a2);

    t = N_NODES * HC;
    k_init<<<(t + B - 1) / B, B, 0, stream>>>(out, emax, z);

    t = N_EDGES * HEADS;
    k_edge_max<<<(t + B - 1) / B, B, 0, stream>>>(src, dst, a1, a2, emax);
    k_edge_acc<<<(t + B - 1) / B, B, 0, stream>>>(src, dst, a1, a2, emax, ft, z, out);

    t = N_NODES * HC;
    k_norm<<<(t + B - 1) / B, B, 0, stream>>>(out, z);
}
